// MoRAPEForCausalLM_81784767251166
// MI455X (gfx1250) — matrix operand order repaired and checked by reading, not compiled, not run on hardware
//
#include <hip/hip_runtime.h>
#include <cstdint>
#include <cstddef>

// ---------------- model dims ----------------
#define Bb 2
#define Ss 2048
#define Dd 1024
#define Hh 16
#define DHh 64
#define Ff 4096
#define Vv 32000
#define NBl 6

typedef __bf16 bf16_t;
typedef __attribute__((ext_vector_type(16))) __bf16 v16bf;
typedef __attribute__((ext_vector_type(8)))  float  v8f;

union FragAB { uint4 u[2]; v16bf v; };

#define WMMA_BF16(a, b, c) \
  __builtin_amdgcn_wmma_f32_16x16x32_bf16(false, (a), false, (b), (short)0, (c), false, false)

// A-fragment (16x32, M x K) from row-major LDS (stride in elements).
// lane&15 = M row; half (lane>>4) selects K runs {8h..8h+7, 16+8h..23+8h}.
__device__ __forceinline__ v16bf frag_a(const bf16_t* base, int lane, int stride) {
  int m = lane & 15;
  int h = (lane >> 4) << 3;  // 0 or 8
  FragAB f;
  f.u[0] = *(const uint4*)(base + m * stride + h);
  f.u[1] = *(const uint4*)(base + m * stride + 16 + h);
  return f.v;
}

// B-fragment (32x16, K x N) from K-major LDS Bt[n][k] (stride elems).
// lane&15 = N col; half (lane>>4) selects K runs {8h..8h+7, 16+8h..23+8h}.
__device__ __forceinline__ v16bf frag_b(const bf16_t* baseT, int lane, int stride) {
  int n = lane & 15;
  int h = (lane >> 4) << 3;  // 0 or 8
  FragAB f;
  f.u[0] = *(const uint4*)(baseT + n * stride + h);
  f.u[1] = *(const uint4*)(baseT + n * stride + 16 + h);
  return f.v;
}

// async global->LDS copy of 32 bytes per lane (two b128 beats; INST_OFFSET
// applies to both the LDS and global address per CDNA5 ISA 10.7).
__device__ __forceinline__ void async_copy32(unsigned ldsAddr, const void* gaddr) {
  asm volatile(
      "global_load_async_to_lds_b128 %0, %1, off\n\t"
      "global_load_async_to_lds_b128 %0, %1, off offset:16"
      :
      : "v"(ldsAddr), "v"((unsigned long long)(uintptr_t)gaddr)
      : "memory");
}
__device__ __forceinline__ void async_wait0() {
  asm volatile("s_wait_asynccnt 0" ::: "memory");
}

// ---------------- GEMM: C[MxN] = A[MxK] @ Bt[NxK]^T, bf16 in, f32 acc ----------------
// B operand is pre-transposed to [N x K] (K-major), so both tiles stage identically.
// mode 0: outF = acc
// mode 1: outF = res + acc          (residual add; outF may alias res)
// mode 2: outB = bf16(acc)
// mode 3: outB = bf16(silu(res) * acc)   (fused SwiGLU gate)
__global__ __launch_bounds__(256) void gemm_kernel(
    const bf16_t* __restrict__ A, const bf16_t* __restrict__ Bt,
    int M, int N, int K,
    const float* res, float* outF, bf16_t* outB, int mode) {
  __shared__ bf16_t sA[128 * 32];
  __shared__ bf16_t sBt[128 * 32];
  int tid = threadIdx.x, lane = tid & 31, w = tid >> 5;
  int m0 = blockIdx.y * 128, n0 = blockIdx.x * 128;
  int rm = (w & 3) * 32, cn = (w >> 2) * 64;

  v8f acc[2][4];
#pragma unroll
  for (int a = 0; a < 2; ++a)
#pragma unroll
    for (int b = 0; b < 4; ++b)
#pragma unroll
      for (int e = 0; e < 8; ++e) acc[a][b][e] = 0.f;

  int row = tid >> 1, half = (tid & 1) * 16;
  unsigned ldsA = (unsigned)(uintptr_t)(sA + row * 32 + half);
  unsigned ldsB = (unsigned)(uintptr_t)(sBt + row * 32 + half);
  const bf16_t* gA = A + (size_t)(m0 + row) * K + half;
  const bf16_t* gB = Bt + (size_t)(n0 + row) * K + half;

  for (int kk = 0; kk < K; kk += 32) {
    async_copy32(ldsA, gA + kk);
    async_copy32(ldsB, gB + kk);
    if (kk + 32 < K) {
      __builtin_prefetch(gA + kk + 32, 0, 1);
      __builtin_prefetch(gB + kk + 32, 0, 1);
    }
    async_wait0();
    __syncthreads();
#pragma unroll
    for (int mi = 0; mi < 2; ++mi) {
      v16bf af = frag_a(sA + (rm + mi * 16) * 32, lane, 32);
#pragma unroll
      for (int ni = 0; ni < 4; ++ni) {
        v16bf bfr = frag_b(sBt + (cn + ni * 16) * 32, lane, 32);
        acc[mi][ni] = WMMA_BF16(af, bfr, acc[mi][ni]);
      }
    }
    __syncthreads();
  }

  int ln = lane & 15, lh = lane >> 4;
#pragma unroll
  for (int mi = 0; mi < 2; ++mi)
#pragma unroll
    for (int ni = 0; ni < 4; ++ni)
#pragma unroll
      for (int j = 0; j < 8; ++j) {
        size_t m = (size_t)m0 + rm + mi * 16 + j + 8 * lh;
        size_t n = (size_t)n0 + cn + ni * 16 + ln;
        size_t idx = m * (size_t)N + n;
        float va = acc[mi][ni][j];
        if (mode == 0) {
          outF[idx] = va;
        } else if (mode == 1) {
          outF[idx] = res[idx] + va;
        } else if (mode == 2) {
          outB[idx] = (bf16_t)va;
        } else {
          float g = res[idx];
          outB[idx] = (bf16_t)((g / (1.f + __expf(-g))) * va);
        }
      }
}

// ---------------- RMSNorm (f32 in, bf16*weight out) ----------------
__global__ __launch_bounds__(256) void rmsnorm_kernel(const float* __restrict__ x,
                                                      const float* __restrict__ w,
                                                      bf16_t* __restrict__ out) {
  __shared__ float red[256];
  size_t t = blockIdx.x;
  const float* xr = x + t * Dd;
  float s = 0.f;
  for (int d = threadIdx.x; d < Dd; d += 256) { float v = xr[d]; s += v * v; }
  red[threadIdx.x] = s;
  __syncthreads();
  for (int o = 128; o > 0; o >>= 1) {
    if (threadIdx.x < o) red[threadIdx.x] += red[threadIdx.x + o];
    __syncthreads();
  }
  float inv = rsqrtf(red[0] / (float)Dd + 1e-6f);
  bf16_t* orow = out + t * Dd;
  for (int d = threadIdx.x; d < Dd; d += 256) orow[d] = (bf16_t)(xr[d] * inv * w[d]);
}

// ---------------- Flash attention: 1 wave per (16 q-rows, head, batch) ----------------
__global__ __launch_bounds__(32) void attn_kernel(const bf16_t* __restrict__ q,
                                                  const bf16_t* __restrict__ k,
                                                  const bf16_t* __restrict__ v,
                                                  bf16_t* __restrict__ o, int L) {
  __shared__ bf16_t sP[16 * 32];
  __shared__ bf16_t sVt[64 * 32];
  int lane = threadIdx.x;
  int q0 = blockIdx.x * 16;
  int hd0 = blockIdx.y * DHh;
  size_t tb = (size_t)blockIdx.z * L;
  int ln = lane & 15, lh = lane >> 4;

  // Q fragments (16x64 split into two 16x32 A-frags)
  const bf16_t* qr = q + (tb + q0 + ln) * Dd + hd0;
  FragAB qa0, qa1;
  {
    int h16 = lh << 4;
    qa0.u[0] = *(const uint4*)(qr + h16);
    qa0.u[1] = *(const uint4*)(qr + h16 + 8);
    qa1.u[0] = *(const uint4*)(qr + 32 + h16);
    qa1.u[1] = *(const uint4*)(qr + 40 + h16);
  }

  float mrow[8], lrow[8];
  v8f oacc[4];
#pragma unroll
  for (int j = 0; j < 8; ++j) { mrow[j] = -1e30f; lrow[j] = 0.f; }
#pragma unroll
  for (int ni = 0; ni < 4; ++ni)
#pragma unroll
    for (int e = 0; e < 8; ++e) oacc[ni][e] = 0.f;

  int nk = (q0 + 16 + 31) / 32;
  for (int c = 0; c < nk; ++c) {
    int kb = c * 32;
    v8f s0, s1;
#pragma unroll
    for (int e = 0; e < 8; ++e) { s0[e] = 0.f; s1[e] = 0.f; }

    int h16 = lh << 4;
    const bf16_t* kr0 = k + (tb + kb + ln) * Dd + hd0;
    const bf16_t* kr1 = k + (tb + kb + 16 + ln) * Dd + hd0;
    FragAB f;
    f.u[0] = *(const uint4*)(kr0 + h16);
    f.u[1] = *(const uint4*)(kr0 + h16 + 8);
    s0 = WMMA_BF16(qa0.v, f.v, s0);
    f.u[0] = *(const uint4*)(kr0 + 32 + h16);
    f.u[1] = *(const uint4*)(kr0 + 40 + h16);
    s0 = WMMA_BF16(qa1.v, f.v, s0);
    f.u[0] = *(const uint4*)(kr1 + h16);
    f.u[1] = *(const uint4*)(kr1 + h16 + 8);
    s1 = WMMA_BF16(qa0.v, f.v, s1);
    f.u[0] = *(const uint4*)(kr1 + 32 + h16);
    f.u[1] = *(const uint4*)(kr1 + 40 + h16);
    s1 = WMMA_BF16(qa1.v, f.v, s1);

    // online softmax per row (rows j+8*lh spread across 16-lane half-wave)
#pragma unroll
    for (int j = 0; j < 8; ++j) {
      int qp = q0 + j + 8 * lh;
      float a0 = (kb + ln <= qp) ? s0[j] * 0.125f : -1e30f;
      float a1 = (kb + 16 + ln <= qp) ? s1[j] * 0.125f : -1e30f;
      float t = fmaxf(a0, a1);
      t = fmaxf(t, __shfl_xor(t, 1, 32));
      t = fmaxf(t, __shfl_xor(t, 2, 32));
      t = fmaxf(t, __shfl_xor(t, 4, 32));
      t = fmaxf(t, __shfl_xor(t, 8, 32));
      float mnew = fmaxf(mrow[j], t);
      float alpha = __expf(mrow[j] - mnew);
      float p0 = __expf(a0 - mnew);
      float p1 = __expf(a1 - mnew);
      float rs = p0 + p1;
      rs += __shfl_xor(rs, 1, 32);
      rs += __shfl_xor(rs, 2, 32);
      rs += __shfl_xor(rs, 4, 32);
      rs += __shfl_xor(rs, 8, 32);
      lrow[j] = lrow[j] * alpha + rs;
      mrow[j] = mnew;
#pragma unroll
      for (int ni = 0; ni < 4; ++ni) oacc[ni][j] *= alpha;
      sP[(j + 8 * lh) * 32 + ln] = (bf16_t)p0;
      sP[(j + 8 * lh) * 32 + 16 + ln] = (bf16_t)p1;
    }

    // stage V chunk (32 tokens x 64 d) transposed into sVt[d][k]
    {
      const bf16_t* vr = v + (tb + kb + lane) * Dd + hd0;
      bf16_t tmp[64];
#pragma unroll
      for (int i = 0; i < 8; ++i) *(uint4*)(tmp + 8 * i) = *(const uint4*)(vr + 8 * i);
#pragma unroll
      for (int dd = 0; dd < 64; ++dd) sVt[dd * 32 + lane] = tmp[dd];
    }
    __syncthreads();

    v16bf pa = frag_a(sP, lane, 32);
#pragma unroll
    for (int ni = 0; ni < 4; ++ni) {
      v16bf bfr = frag_b(sVt + (ni * 16) * 32, lane, 32);
      oacc[ni] = WMMA_BF16(pa, bfr, oacc[ni]);
    }
    __syncthreads();
  }

#pragma unroll
  for (int ni = 0; ni < 4; ++ni)
#pragma unroll
    for (int j = 0; j < 8; ++j) {
      size_t rowt = tb + q0 + j + 8 * lh;
      o[rowt * Dd + hd0 + ni * 16 + ln] = (bf16_t)(oacc[ni][j] / lrow[j]);
    }
}

// ---------------- router: probs = ALPHA * sigmoid(x . rw) ----------------
__global__ __launch_bounds__(256) void router_kernel(const float* __restrict__ x,
                                                     const float* __restrict__ rw,
                                                     float* __restrict__ probs) {
  __shared__ float red[256];
  size_t t = blockIdx.x;
  const float* xr = x + t * Dd;
  float s = 0.f;
  for (int d = threadIdx.x; d < Dd; d += 256) s += xr[d] * rw[d];
  red[threadIdx.x] = s;
  __syncthreads();
  for (int o = 128; o > 0; o >>= 1) {
    if (threadIdx.x < o) red[threadIdx.x] += red[threadIdx.x + o];
    __syncthreads();
  }
  if (threadIdx.x == 0) probs[t] = 0.1f / (1.f + __expf(-red[0]));
}

// ---------------- top-k via bitonic sort (per batch row) ----------------
#define NSORT 2048
__global__ __launch_bounds__(1024) void topk_kernel(const float* __restrict__ probs, int L, int Kk,
                                                    const int* prev, int PL,
                                                    int* __restrict__ topi, float* __restrict__ topw,
                                                    int* __restrict__ orig) {
  __shared__ float skey[NSORT];
  __shared__ int sidx[NSORT];
  int b = blockIdx.x, tid = threadIdx.x;
  for (int i = tid; i < NSORT; i += 1024) {
    if (i < L) { skey[i] = probs[(size_t)b * L + i]; sidx[i] = i; }
    else       { skey[i] = -1e38f; sidx[i] = 0x7fffffff; }
  }
  __syncthreads();
  // sort by (prob desc, idx asc) -- matches jax.lax.top_k tie-breaking
  for (int size = 2; size <= NSORT; size <<= 1)
    for (int stride = size >> 1; stride > 0; stride >>= 1) {
      for (int i = tid; i < NSORT; i += 1024) {
        int j = i ^ stride;
        if (j > i) {
          bool asc = ((i & size) == 0);
          float ki = skey[i], kj = skey[j];
          int ii = sidx[i], ij = sidx[j];
          bool iFirst = (ki > kj) || (ki == kj && ii < ij);
          if (asc ? !iFirst : iFirst) { skey[i] = kj; skey[j] = ki; sidx[i] = ij; sidx[j] = ii; }
        }
      }
      __syncthreads();
    }
  // keep first Kk, push the rest to the end, re-sort by index ascending
  for (int i = tid; i < NSORT; i += 1024)
    if (i >= Kk) sidx[i] = 0x7fffffff;
  __syncthreads();
  for (int size = 2; size <= NSORT; size <<= 1)
    for (int stride = size >> 1; stride > 0; stride >>= 1) {
      for (int i = tid; i < NSORT; i += 1024) {
        int j = i ^ stride;
        if (j > i) {
          bool asc = ((i & size) == 0);
          int ii = sidx[i], ij = sidx[j];
          bool iFirst = (ii < ij);
          if (asc ? !iFirst : iFirst) {
            float ki = skey[i];
            skey[i] = skey[j]; skey[j] = ki;
            sidx[i] = ij; sidx[j] = ii;
          }
        }
      }
      __syncthreads();
    }
  for (int i = tid; i < Kk; i += 1024) {
    int id = sidx[i];
    topi[(size_t)b * Kk + i] = id;
    topw[(size_t)b * Kk + i] = skey[i];
    orig[(size_t)b * Kk + i] = prev ? prev[(size_t)b * PL + id] : id;
  }
}

// ---------------- gather / scatter / embed / converts ----------------
__global__ __launch_bounds__(256) void gather_kernel(const float* __restrict__ src,
                                                     const int* __restrict__ ind,
                                                     float* __restrict__ dst, int Kk, int Lsrc) {
  int b = blockIdx.y, j = blockIdx.x;
  const float* s = src + ((size_t)b * Lsrc + ind[(size_t)b * Kk + j]) * Dd;
  float* d = dst + ((size_t)b * Kk + j) * Dd;
  for (int i = threadIdx.x; i < Dd; i += 256) d[i] = s[i];
}

__global__ __launch_bounds__(256) void scatter_kernel(float* h, const float* __restrict__ sel,
                                                      const int* __restrict__ orig,
                                                      const float* __restrict__ w, int Kk, int Lh) {
  int b = blockIdx.y, j = blockIdx.x;
  float* d = h + ((size_t)b * Lh + orig[(size_t)b * Kk + j]) * Dd;
  const float* s = sel + ((size_t)b * Kk + j) * Dd;
  float ww = w[(size_t)b * Kk + j];
  for (int i = threadIdx.x; i < Dd; i += 256) d[i] += s[i] * ww;
}

__global__ __launch_bounds__(256) void embed_kernel(const int* __restrict__ ids,
                                                    const float* __restrict__ emb,
                                                    const float* __restrict__ pos,
                                                    float* __restrict__ h) {
  size_t t = blockIdx.x;
  int s = (int)(t % Ss);
  size_t id = (size_t)ids[t];
  for (int i = threadIdx.x; i < Dd; i += 256)
    h[t * Dd + i] = emb[id * Dd + i] + pos[(size_t)s * Dd + i];
}

__global__ __launch_bounds__(256) void cvt_kernel(const float* __restrict__ in,
                                                  bf16_t* __restrict__ out, size_t n) {
  size_t i = (size_t)blockIdx.x * 256 + threadIdx.x;
  if (i < n) out[i] = (bf16_t)in[i];
}

// in [R x C] f32 -> out [C x R] bf16 (pre-transpose weights to K-major)
__global__ __launch_bounds__(256) void transpose_cvt_kernel(const float* __restrict__ in,
                                                            bf16_t* __restrict__ out,
                                                            int R, int C) {
  __shared__ float tile[32][33];
  int rb = blockIdx.y * 32, cb = blockIdx.x * 32;
  int tx = threadIdx.x & 31, ty = threadIdx.x >> 5;  // 32 x 8
  for (int r = ty; r < 32; r += 8) tile[r][tx] = in[(size_t)(rb + r) * C + cb + tx];
  __syncthreads();
  for (int r = ty; r < 32; r += 8) out[(size_t)(cb + r) * R + rb + tx] = (bf16_t)tile[tx][r];
}

// ---------------- host orchestration ----------------
extern "C" void kernel_launch(void* const* d_in, const int* in_sizes, int n_in,
                              void* d_out, int out_size, void* d_ws, size_t ws_size,
                              hipStream_t stream) {
  (void)in_sizes; (void)n_in; (void)out_size; (void)ws_size;
  const int* ids = (const int*)d_in[0];
  const float* embed = (const float*)d_in[1];
  const float* pos = (const float*)d_in[2];
  const float* Wq = (const float*)d_in[3];
  const float* Wk = (const float*)d_in[4];
  const float* Wv = (const float*)d_in[5];
  const float* Wo = (const float*)d_in[6];
  const float* Wg = (const float*)d_in[7];
  const float* Wu = (const float*)d_in[8];
  const float* Wd = (const float*)d_in[9];
  const float* ln1 = (const float*)d_in[10];
  const float* ln2 = (const float*)d_in[11];
  const float* router = (const float*)d_in[12];
  const float* fnorm = (const float*)d_in[13];

  char* ws = (char*)d_ws;
  size_t off = 0;
  auto alloc = [&](size_t bytes) -> void* {
    void* p = ws + off;
    off = (off + bytes + 255) & ~(size_t)255;
    return p;
  };

  const size_t DD = (size_t)Dd * Dd, DF = (size_t)Dd * Ff;
  // all weight copies stored TRANSPOSED: [N x K], K contiguous
  bf16_t* wqb = (bf16_t*)alloc(NBl * DD * 2);
  bf16_t* wkb = (bf16_t*)alloc(NBl * DD * 2);
  bf16_t* wvb = (bf16_t*)alloc(NBl * DD * 2);
  bf16_t* wob = (bf16_t*)alloc(NBl * DD * 2);
  bf16_t* wgb = (bf16_t*)alloc(NBl * DF * 2);
  bf16_t* wub = (bf16_t*)alloc(NBl * DF * 2);
  bf16_t* wdb = (bf16_t*)alloc(NBl * DF * 2);
  bf16_t* embB = (bf16_t*)alloc((size_t)Vv * Dd * 2);  // lm-head Bt = embed itself

  const size_t RT = (size_t)Bb * Ss;  // 4096 max rows
  float* Hbuf = (float*)alloc(RT * Dd * 4);
  bf16_t* XN = (bf16_t*)alloc(RT * Dd * 2);
  bf16_t* QB = (bf16_t*)alloc(RT * Dd * 2);
  bf16_t* KB = (bf16_t*)alloc(RT * Dd * 2);
  bf16_t* VB = (bf16_t*)alloc(RT * Dd * 2);
  bf16_t* OB = (bf16_t*)alloc(RT * Dd * 2);
  float* GB = (float*)alloc(RT * Ff * 4);
  bf16_t* GU = (bf16_t*)alloc(RT * Ff * 2);
  float* SEL = (float*)alloc((size_t)Bb * 1024 * Dd * 4);
  float* X1 = (float*)alloc((size_t)Bb * 1024 * Dd * 4);
  float* PROBS = (float*)alloc((size_t)Bb * 2048 * 4);
  int* TOPI0 = (int*)alloc(Bb * 1024 * 4);
  float* TOPW0 = (float*)alloc(Bb * 1024 * 4);
  int* ORIG0 = (int*)alloc(Bb * 1024 * 4);
  int* TOPI1 = (int*)alloc(Bb * 512 * 4);
  float* TOPW1 = (float*)alloc(Bb * 512 * 4);
  int* ORIG1 = (int*)alloc(Bb * 512 * 4);

  // weight preprocessing: transpose+convert each [K x N] block to [N x K] bf16
  auto cvtT = [&](const float* src, bf16_t* dst, int R, int C) {
    transpose_cvt_kernel<<<dim3(C / 32, R / 32), 256, 0, stream>>>(src, dst, R, C);
  };
  for (int bi = 0; bi < NBl; ++bi) {
    cvtT(Wq + bi * DD, wqb + bi * DD, Dd, Dd);
    cvtT(Wk + bi * DD, wkb + bi * DD, Dd, Dd);
    cvtT(Wv + bi * DD, wvb + bi * DD, Dd, Dd);
    cvtT(Wo + bi * DD, wob + bi * DD, Dd, Dd);
    cvtT(Wg + bi * DF, wgb + bi * DF, Dd, Ff);
    cvtT(Wu + bi * DF, wub + bi * DF, Dd, Ff);
    cvtT(Wd + bi * DF, wdb + bi * DF, Ff, Dd);
  }
  {
    size_t n = (size_t)Vv * Dd;
    cvt_kernel<<<(unsigned)((n + 255) / 256), 256, 0, stream>>>(embed, embB, n);
  }

  auto gemm = [&](const bf16_t* A, const bf16_t* Bt, int M, int N, int K,
                  const float* res, float* oF, bf16_t* oB, int mode) {
    gemm_kernel<<<dim3(N / 128, M / 128), 256, 0, stream>>>(A, Bt, M, N, K, res, oF, oB, mode);
  };

  auto run_block = [&](float* x, int L, int bi) {
    int R = Bb * L;
    rmsnorm_kernel<<<R, 256, 0, stream>>>(x, ln1 + (size_t)bi * Dd, XN);
    gemm(XN, wqb + (size_t)bi * DD, R, Dd, Dd, nullptr, nullptr, QB, 2);
    gemm(XN, wkb + (size_t)bi * DD, R, Dd, Dd, nullptr, nullptr, KB, 2);
    gemm(XN, wvb + (size_t)bi * DD, R, Dd, Dd, nullptr, nullptr, VB, 2);
    attn_kernel<<<dim3(L / 16, Hh, Bb), 32, 0, stream>>>(QB, KB, VB, OB, L);
    gemm(OB, wob + (size_t)bi * DD, R, Dd, Dd, x, x, nullptr, 1);
    rmsnorm_kernel<<<R, 256, 0, stream>>>(x, ln2 + (size_t)bi * Dd, XN);
    gemm(XN, wgb + (size_t)bi * DF, R, Ff, Dd, nullptr, GB, nullptr, 0);
    gemm(XN, wub + (size_t)bi * DF, R, Ff, Dd, GB, nullptr, GU, 3);
    gemm(GU, wdb + (size_t)bi * DF, R, Dd, Ff, x, x, nullptr, 1);
  };

  // embedding + positions
  embed_kernel<<<(unsigned)(Bb * Ss), 256, 0, stream>>>(ids, embed, pos, Hbuf);

  // shared block 0 (full length)
  run_block(Hbuf, Ss, 0);

  // recursion 0: L=2048, k=1024
  router_kernel<<<(unsigned)(Bb * 2048), 256, 0, stream>>>(Hbuf, router, PROBS);
  topk_kernel<<<Bb, 1024, 0, stream>>>(PROBS, 2048, 1024, nullptr, 0, TOPI0, TOPW0, ORIG0);
  gather_kernel<<<dim3(1024, Bb), 256, 0, stream>>>(Hbuf, ORIG0, SEL, 1024, Ss);
  run_block(SEL, 1024, 1);
  run_block(SEL, 1024, 2);
  scatter_kernel<<<dim3(1024, Bb), 256, 0, stream>>>(Hbuf, SEL, ORIG0, TOPW0, 1024, Ss);

  // recursion 1: gather prev tokens, L=1024, k=512, chained indices
  gather_kernel<<<dim3(1024, Bb), 256, 0, stream>>>(Hbuf, ORIG0, X1, 1024, Ss);
  router_kernel<<<(unsigned)(Bb * 1024), 256, 0, stream>>>(X1, router + Dd, PROBS);
  topk_kernel<<<Bb, 1024, 0, stream>>>(PROBS, 1024, 512, ORIG0, 1024, TOPI1, TOPW1, ORIG1);
  gather_kernel<<<dim3(512, Bb), 256, 0, stream>>>(Hbuf, ORIG1, SEL, 512, Ss);
  run_block(SEL, 512, 3);
  run_block(SEL, 512, 4);
  scatter_kernel<<<dim3(512, Bb), 256, 0, stream>>>(Hbuf, SEL, ORIG1, TOPW1, 512, Ss);

  // shared block 5
  run_block(Hbuf, Ss, 5);

  // final norm + tied lm head (Bt = embed [V x D], K-major already)
  rmsnorm_kernel<<<(unsigned)(Bb * Ss), 256, 0, stream>>>(Hbuf, fnorm, XN);
  gemm(XN, embB, Bb * Ss, Vv, Dd, nullptr, (float*)d_out, nullptr, 0);
}
